// BackboneGNN_25941602468402
// MI455X (gfx1250) — compile-verified, matrix operand order checked by reading, not run on hardware
//
#include <hip/hip_runtime.h>
#include <hip/hip_bf16.h>

typedef __attribute__((ext_vector_type(16))) _Float16 v16h;
typedef __attribute__((ext_vector_type(8)))  _Float16 v8h;
typedef __attribute__((ext_vector_type(8)))  float    v8f;

#define RS_    0.7071f
#define SCALE_ 60.0f
#define EPS_   1e-6f

__device__ __forceinline__ float gelu_t(float x) {
    const float c = 0.7978845608028654f; // sqrt(2/pi)
    float t = c * (x + 0.044715f * x * x * x);
    float e = __expf(2.0f * t);          // tanh(t) = (e-1)/(e+1)
    return 0.5f * x * (1.0f + (e - 1.0f) / (e + 1.0f));
}

// ---- CDNA5 async global->LDS copy (16B per lane), tracked by ASYNCcnt -------
__device__ __forceinline__ void async_cp16(_Float16* lds, const _Float16* g) {
    unsigned lofs = (unsigned)(unsigned long long)lds; // LDS aperture: low 32 bits = DS offset
    asm volatile("global_load_async_to_lds_b128 %0, %1, off"
                 :: "v"(lofs), "v"(g) : "memory");
}
__device__ __forceinline__ void wait_async0() {
    asm volatile("s_wait_asynccnt 0x0" ::: "memory");
}

// A-fragment: 16x32 f16 tile from LDS (row-major, lda halves, lda % 8 == 0)
__device__ __forceinline__ v16h ldsA(const _Float16* A, int lda, int mbase, int kk, int lane) {
    const _Float16* p = A + (mbase + (lane & 15)) * lda + kk + ((lane >> 4) << 4);
    v8h lo = *(const v8h*)p;
    v8h hi = *(const v8h*)(p + 8);
    v16h r;
#pragma unroll
    for (int i = 0; i < 8; ++i) { r[i] = lo[i]; r[i + 8] = hi[i]; }
    return r;
}

// B-fragment: 32x16 f16 tile from global weights stored transposed [N][Kd]
__device__ __forceinline__ v16h gblB(const _Float16* Wt, int Kd, int col, int kk, int lane) {
    const _Float16* p = Wt + (col + (lane & 15)) * Kd + kk + ((lane >> 4) << 4);
    return *(const v16h*)p;
}

template <int KD>
__device__ __forceinline__ void wmma_gemm(const _Float16* Alds, int lda,
                                          const _Float16* __restrict__ Wt,
                                          int wave, int lane, v8f acc[4]) {
    const int mbase = (wave & 1) * 16;
    const int nblk  = (wave >> 1) * 64;
    v8f z = {};
    acc[0] = z; acc[1] = z; acc[2] = z; acc[3] = z;
    for (int kk = 0; kk < KD; kk += 32) {
        v16h a = ldsA(Alds, lda, mbase, kk, lane);
#pragma unroll
        for (int nt = 0; nt < 4; ++nt) {
            v16h b = gblB(Wt, KD, nblk + nt * 16, kk, lane);
            acc[nt] = __builtin_amdgcn_wmma_f32_16x16x32_f16(
                false, a, false, b, (short)0, acc[nt], false, false);
        }
    }
}

__device__ __forceinline__ void store_f16(const v8f acc[4], _Float16* Olds, int ldo,
                                          const float* __restrict__ bias, bool do_gelu,
                                          int wave, int lane) {
    const int mbase = (wave & 1) * 16;
    const int nblk  = (wave >> 1) * 64;
    const int rsel  = (lane >> 4) * 8;
#pragma unroll
    for (int nt = 0; nt < 4; ++nt) {
        int col = nblk + nt * 16 + (lane & 15);
        float bv = bias[col];
#pragma unroll
        for (int r = 0; r < 8; ++r) {
            float v = acc[nt][r] + bv;
            if (do_gelu) v = gelu_t(v);
            Olds[(mbase + rsel + r) * ldo + col] = (_Float16)v;
        }
    }
}

__device__ __forceinline__ void store_f32(const v8f acc[4], float* Olds, int ldo,
                                          const float* __restrict__ bias,
                                          int wave, int lane) {
    const int mbase = (wave & 1) * 16;
    const int nblk  = (wave >> 1) * 64;
    const int rsel  = (lane >> 4) * 8;
#pragma unroll
    for (int nt = 0; nt < 4; ++nt) {
        int col = nblk + nt * 16 + (lane & 15);
        float bv = bias[col];
#pragma unroll
        for (int r = 0; r < 8; ++r)
            Olds[(mbase + rsel + r) * ldo + col] = acc[nt][r] + bv;
    }
}

// ---------------- prep: weight [Kd][Nd] f32 -> [Nd][Kd] f16 ------------------
__global__ void wconv_kernel(const float* __restrict__ W, _Float16* __restrict__ Wt,
                             int Kd, int Nd) {
    int idx = blockIdx.x * 256 + threadIdx.x;
    if (idx < Kd * Nd) {
        int k = idx / Nd, n = idx - k * Nd;
        Wt[n * Kd + k] = (_Float16)W[idx];
    }
}
// ---------------- prep: f32 -> f16 elementwise -------------------------------
__global__ void cvt16_kernel(const float* __restrict__ X, _Float16* __restrict__ Y, int n) {
    int i = blockIdx.x * 256 + threadIdx.x;
    if (i < n) Y[i] = (_Float16)X[i];
}

// ---------------- phase A: node message MLP + sum over K + residual + LN -----
__global__ __launch_bounds__(128) void node_msg_kernel(
    const float* __restrict__ hV,        // f32 layer-input h_V (residual)
    const _Float16* __restrict__ hVf,    // f16 mirror of layer-input h_V
    const _Float16* __restrict__ hEf,    // f16 mirror of layer-input h_E
    const int* __restrict__ topo,
    const _Float16* __restrict__ W1t, const float* __restrict__ b1,
    const _Float16* __restrict__ W2t, const float* __restrict__ b2,
    const _Float16* __restrict__ W3t, const float* __restrict__ b3,
    const float* __restrict__ lns, const float* __restrict__ lnb,
    float* __restrict__ hVout, _Float16* __restrict__ hVoutF) {
    __shared__ _Float16 A0[32 * 392];
    __shared__ _Float16 A1[32 * 136];
    __shared__ _Float16 A2[32 * 136];
    __shared__ float outb[32 * 132];
    __shared__ float red[128];
    __shared__ int topoS[32];
    const int tid = threadIdx.x, wave = tid >> 5, lane = tid & 31;
    const int blk = blockIdx.x;        // b*N + i
    const int b = blk >> 11;           // N = 2048
    const int nodeBase = blk * 128;
    if (tid < 32) topoS[tid] = topo[blk * 32 + tid];
    __syncthreads();
    // async-stage A0: 32 rows x 768B = 1536 16B chunks, 12/thread, no VGPR round-trip
    for (int ch = tid; ch < 32 * 48; ch += 128) {
        int r = ch / 48, c = ch - r * 48;      // c = 16B chunk within row
        const _Float16* src;
        if (c < 16)      src = hVf + nodeBase + c * 8;
        else if (c < 32) src = hVf + ((b << 11) + topoS[r]) * 128 + (c - 16) * 8;
        else             src = hEf + (blk * 32 + r) * 128 + (c - 32) * 8;
        async_cp16(&A0[r * 392 + c * 8], src);
    }
    wait_async0();
    __syncthreads();
    v8f acc[4];
    wmma_gemm<384>(A0, 392, W1t, wave, lane, acc);
    store_f16(acc, A1, 136, b1, true, wave, lane);
    __syncthreads();
    wmma_gemm<128>(A1, 136, W2t, wave, lane, acc);
    store_f16(acc, A2, 136, b2, true, wave, lane);
    __syncthreads();
    wmma_gemm<128>(A2, 136, W3t, wave, lane, acc);
    store_f32(acc, outb, 132, b3, wave, lane);
    __syncthreads();
    float s = 0.f;
#pragma unroll 8
    for (int r = 0; r < 32; ++r) s += outb[r * 132 + tid];
    float x = RS_ * hV[nodeBase + tid] + s * (1.0f / SCALE_);
    red[tid] = x; __syncthreads();
    for (int st = 64; st > 0; st >>= 1) { if (tid < st) red[tid] += red[tid + st]; __syncthreads(); }
    float mu = red[0] * (1.0f / 128.0f); __syncthreads();
    float d = x - mu;
    red[tid] = d * d; __syncthreads();
    for (int st = 64; st > 0; st >>= 1) { if (tid < st) red[tid] += red[tid + st]; __syncthreads(); }
    float var = red[0] * (1.0f / 128.0f);
    float y = d * rsqrtf(var + EPS_) * lns[tid] + lnb[tid];
    hVout[nodeBase + tid] = y;
    hVoutF[nodeBase + tid] = (_Float16)y;
}

// ---------------- phase B: position-wise FF + residual + LN ------------------
__global__ __launch_bounds__(128) void ff_kernel(
    const float* __restrict__ hVin, const _Float16* __restrict__ hVinF,
    const _Float16* __restrict__ W1t, const float* __restrict__ b1,
    const _Float16* __restrict__ W2t, const float* __restrict__ b2,
    const float* __restrict__ lns, const float* __restrict__ lnb,
    float* __restrict__ hVout, _Float16* __restrict__ hVoutF) {
    __shared__ _Float16 A0[32 * 136];
    __shared__ _Float16 A1[32 * 136];
    __shared__ float outb[32 * 132];
    __shared__ float ps[32][4], ps2[32][4], muv[32], vav[32];
    const int tid = threadIdx.x, wave = tid >> 5, lane = tid & 31;
    const int base = blockIdx.x * 32 * 128;
    for (int ch = tid; ch < 32 * 16; ch += 128) {
        int r = ch >> 4, c = ch & 15;
        async_cp16(&A0[r * 136 + c * 8], hVinF + base + r * 128 + c * 8);
    }
    wait_async0();
    __syncthreads();
    v8f acc[4];
    wmma_gemm<128>(A0, 136, W1t, wave, lane, acc);
    store_f16(acc, A1, 136, b1, true, wave, lane);
    __syncthreads();
    wmma_gemm<128>(A1, 136, W2t, wave, lane, acc);
    store_f32(acc, outb, 132, b2, wave, lane);
    __syncthreads();
    for (int idx = tid; idx < 32 * 128; idx += 128) {
        int r = idx >> 7, c = idx & 127;
        outb[r * 132 + c] = RS_ * hVin[base + r * 128 + c] + outb[r * 132 + c];
    }
    __syncthreads();
    int row = tid >> 2, sub = tid & 3;
    float s = 0.f, s2 = 0.f;
    for (int c = sub * 32; c < sub * 32 + 32; ++c) { float v = outb[row * 132 + c]; s += v; s2 += v * v; }
    ps[row][sub] = s; ps2[row][sub] = s2; __syncthreads();
    if (sub == 0) {
        float ts = ps[row][0] + ps[row][1] + ps[row][2] + ps[row][3];
        float t2 = ps2[row][0] + ps2[row][1] + ps2[row][2] + ps2[row][3];
        float mu = ts * (1.0f / 128.0f);
        muv[row] = mu; vav[row] = t2 * (1.0f / 128.0f) - mu * mu;
    }
    __syncthreads();
    float mu = muv[row], rstd = rsqrtf(vav[row] + EPS_);
    for (int c = sub * 32; c < sub * 32 + 32; ++c) {
        float v = outb[row * 132 + c];
        float y = (v - mu) * rstd * lns[c] + lnb[c];
        hVout[base + row * 128 + c] = y;
        hVoutF[base + row * 128 + c] = (_Float16)y;
    }
}

// ---------------- phase C: edge MLP + residual + per-edge LN (in-place) ------
__global__ __launch_bounds__(128) void edge_kernel(
    const _Float16* __restrict__ hVf,    // f16 updated h_V (post-FF)
    _Float16* __restrict__ hEf,          // f16 h_E mirror, updated in place
    const float* __restrict__ hEr,       // f32 h_E for residual (layer input)
    const int* __restrict__ topo,
    const _Float16* __restrict__ W1t, const float* __restrict__ b1,
    const _Float16* __restrict__ W2t, const float* __restrict__ b2,
    const _Float16* __restrict__ W3t, const float* __restrict__ b3,
    const float* __restrict__ lns, const float* __restrict__ lnb,
    float* __restrict__ hEout) {
    __shared__ _Float16 A0[32 * 392];
    __shared__ _Float16 A1[32 * 136];
    __shared__ _Float16 A2[32 * 136];
    __shared__ float outb[32 * 132];
    __shared__ int topoS[32];
    __shared__ float ps[32][4], ps2[32][4], muv[32], vav[32];
    const int tid = threadIdx.x, wave = tid >> 5, lane = tid & 31;
    const int blk = blockIdx.x;
    const int b = blk >> 11;
    const int nodeBase = blk * 128;
    if (tid < 32) topoS[tid] = topo[blk * 32 + tid];
    __syncthreads();
    for (int ch = tid; ch < 32 * 48; ch += 128) {
        int r = ch / 48, c = ch - r * 48;
        const _Float16* src;
        if (c < 16)      src = hVf + nodeBase + c * 8;
        else if (c < 32) src = hVf + ((b << 11) + topoS[r]) * 128 + (c - 16) * 8;
        else             src = hEf + (blk * 32 + r) * 128 + (c - 32) * 8;
        async_cp16(&A0[r * 392 + c * 8], src);
    }
    wait_async0();
    __syncthreads();
    v8f acc[4];
    wmma_gemm<384>(A0, 392, W1t, wave, lane, acc);
    store_f16(acc, A1, 136, b1, true, wave, lane);
    __syncthreads();
    wmma_gemm<128>(A1, 136, W2t, wave, lane, acc);
    store_f16(acc, A2, 136, b2, true, wave, lane);
    __syncthreads();
    wmma_gemm<128>(A2, 136, W3t, wave, lane, acc);
    store_f32(acc, outb, 132, b3, wave, lane);
    __syncthreads();
    for (int idx = tid; idx < 32 * 128; idx += 128) {
        int r = idx >> 7, c = idx & 127;
        outb[r * 132 + c] = RS_ * hEr[(blk * 32 + r) * 128 + c] + outb[r * 132 + c];
    }
    __syncthreads();
    int row = tid >> 2, sub = tid & 3;
    float s = 0.f, s2 = 0.f;
    for (int c = sub * 32; c < sub * 32 + 32; ++c) { float v = outb[row * 132 + c]; s += v; s2 += v * v; }
    ps[row][sub] = s; ps2[row][sub] = s2; __syncthreads();
    if (sub == 0) {
        float ts = ps[row][0] + ps[row][1] + ps[row][2] + ps[row][3];
        float t2 = ps2[row][0] + ps2[row][1] + ps2[row][2] + ps2[row][3];
        float mu = ts * (1.0f / 128.0f);
        muv[row] = mu; vav[row] = t2 * (1.0f / 128.0f) - mu * mu;
    }
    __syncthreads();
    float mu = muv[row], rstd = rsqrtf(vav[row] + EPS_);
    for (int c = sub * 32; c < sub * 32 + 32; ++c) {
        float v = outb[row * 132 + c];
        float y = (v - mu) * rstd * lns[c] + lnb[c];
        hEout[(blk * 32 + row) * 128 + c] = y;
        hEf  [(blk * 32 + row) * 128 + c] = (_Float16)y;
    }
}

extern "C" void kernel_launch(void* const* d_in, const int* in_sizes, int n_in,
                              void* d_out, int out_size, void* d_ws, size_t ws_size,
                              hipStream_t stream) {
    (void)in_sizes; (void)n_in; (void)out_size; (void)ws_size;
    const float* hV  = (const float*)d_in[0];
    const float* hE  = (const float*)d_in[1];
    const int*  topo = (const int*)d_in[2];
    const float* nw1 = (const float*)d_in[3];  const float* nb1 = (const float*)d_in[4];
    const float* nw2 = (const float*)d_in[5];  const float* nb2 = (const float*)d_in[6];
    const float* nw3 = (const float*)d_in[7];  const float* nb3 = (const float*)d_in[8];
    const float* fw1 = (const float*)d_in[9];  const float* fb1 = (const float*)d_in[10];
    const float* fw2 = (const float*)d_in[11]; const float* fb2 = (const float*)d_in[12];
    const float* ew1 = (const float*)d_in[13]; const float* eb1 = (const float*)d_in[14];
    const float* ew2 = (const float*)d_in[15]; const float* eb2 = (const float*)d_in[16];
    const float* ew3 = (const float*)d_in[17]; const float* eb3 = (const float*)d_in[18];
    const float* l1s = (const float*)d_in[19]; const float* l1b = (const float*)d_in[20];
    const float* l2s = (const float*)d_in[21]; const float* l2b = (const float*)d_in[22];
    const float* l3s = (const float*)d_in[23]; const float* l3b = (const float*)d_in[24];

    const int NV = 2 * 2048 * 128;            // 524288
    const int NE = 2 * 2048 * 32 * 128;       // 16777216
    float* hVout = (float*)d_out;
    float* hEout = hVout + (size_t)NV;

    char* ws = (char*)d_ws;
    _Float16* wts   = (_Float16*)ws;                      // 3*196608 halves (~1.2 MB)
    float*    hv1   = (float*)(ws + ((size_t)2 << 20));   // 2 MB f32
    _Float16* hvinF = (_Float16*)(ws + ((size_t)4 << 20));// 1 MB f16 (layer-input/updated h_V)
    _Float16* hv1F  = (_Float16*)(ws + ((size_t)5 << 20));// 1 MB f16 (post-phase-A h_V)
    _Float16* hEF   = (_Float16*)(ws + ((size_t)6 << 20));// 32 MB f16 h_E mirror

    const size_t LSTRIDE = 196608;
    for (int l = 0; l < 3; ++l) {
        _Float16* base = wts + (size_t)l * LSTRIDE;
        wconv_kernel<<<(384 * 128 + 255) / 256, 256, 0, stream>>>(nw1 + (size_t)l * 384 * 128, base + 0,      384, 128);
        wconv_kernel<<<(128 * 128 + 255) / 256, 256, 0, stream>>>(nw2 + (size_t)l * 128 * 128, base + 49152,  128, 128);
        wconv_kernel<<<(128 * 128 + 255) / 256, 256, 0, stream>>>(nw3 + (size_t)l * 128 * 128, base + 65536,  128, 128);
        wconv_kernel<<<(128 * 128 + 255) / 256, 256, 0, stream>>>(fw1 + (size_t)l * 128 * 128, base + 81920,  128, 128);
        wconv_kernel<<<(128 * 128 + 255) / 256, 256, 0, stream>>>(fw2 + (size_t)l * 128 * 128, base + 98304,  128, 128);
        wconv_kernel<<<(384 * 128 + 255) / 256, 256, 0, stream>>>(ew1 + (size_t)l * 384 * 128, base + 114688, 384, 128);
        wconv_kernel<<<(128 * 128 + 255) / 256, 256, 0, stream>>>(ew2 + (size_t)l * 128 * 128, base + 163840, 128, 128);
        wconv_kernel<<<(128 * 128 + 255) / 256, 256, 0, stream>>>(ew3 + (size_t)l * 128 * 128, base + 180224, 128, 128);
    }
    cvt16_kernel<<<(NV + 255) / 256, 256, 0, stream>>>(hV, hvinF, NV);
    cvt16_kernel<<<(NE + 255) / 256, 256, 0, stream>>>(hE, hEF, NE);

    for (int l = 0; l < 3; ++l) {
        _Float16* base = wts + (size_t)l * LSTRIDE;
        const float* hv_in = (l == 0) ? hV : hVout;       // f32 residual source
        const float* heR   = (l == 0) ? hE : hEout;       // f32 edge residual source
        // phase A: hv1 = LN(RS*hV + sum_k nodeMLP)/...  (writes f32 + f16 mirror)
        node_msg_kernel<<<4096, 128, 0, stream>>>(hv_in, hvinF, hEF, topo,
            base + 0, nb1 + l * 128, base + 49152, nb2 + l * 128, base + 65536, nb3 + l * 128,
            l1s + l * 128, l1b + l * 128, hv1, hv1F);
        // phase B: hVout = LN(RS*hv1 + FF(hv1)); f16 mirror into hvinF (next layer input)
        ff_kernel<<<128, 128, 0, stream>>>(hv1, hv1F,
            base + 81920, fb1 + l * 128, base + 98304, fb2 + l * 128,
            l2s + l * 128, l2b + l * 128, hVout, hvinF);
        // phase C: h_E update in place (f32 in d_out, f16 mirror)
        edge_kernel<<<4096, 128, 0, stream>>>(hvinF, hEF, heR, topo,
            base + 114688, eb1 + l * 128, base + 163840, eb2 + l * 128, base + 180224, eb3 + l * 128,
            l3s + l * 128, l3b + l * 128, hEout);
    }
}
